// ContrastiveLoss_61426622267469
// MI455X (gfx1250) — matrix operand order repaired and checked by reading, not compiled, not run on hardware
//
#include <hip/hip_runtime.h>
#include <math.h>

// ---------------------------------------------------------------------------
// Problem constants (reference: N=4096 rows, D=1024 features, fp32)
// ---------------------------------------------------------------------------
#define NROWS 4096
#define DIM   1024
#define MARGIN 0.2f
#define BALANCE 0.5f
#define EPS 1e-18f

typedef __attribute__((ext_vector_type(16))) __bf16 v16bf;
typedef __attribute__((ext_vector_type(8)))  __bf16 v8bf;
typedef __attribute__((ext_vector_type(4)))  __bf16 v4bf;
typedef __attribute__((ext_vector_type(8)))  float  v8f;

// ---------------------------------------------------------------------------
// Zero the row/col accumulators (graph-safe, no memset)
// ---------------------------------------------------------------------------
__global__ void zero_kernel(float* __restrict__ p, int n) {
    int i = blockIdx.x * blockDim.x + threadIdx.x;
    if (i < n) p[i] = 0.0f;
}

// ---------------------------------------------------------------------------
// Row L2-normalize fp32 -> bf16.  One block per row, 256 threads x 4 floats.
// ---------------------------------------------------------------------------
__global__ __launch_bounds__(256)
void norm_bf16_kernel(const float* __restrict__ src, __bf16* __restrict__ dst) {
    __shared__ float red[8];
    const int row = blockIdx.x;
    const int t   = threadIdx.x;
    const float4* s4 = (const float4*)(src + (size_t)row * DIM);
    float4 x = s4[t];
    float ss = x.x * x.x + x.y * x.y + x.z * x.z + x.w * x.w;
#pragma unroll
    for (int m = 16; m >= 1; m >>= 1) ss += __shfl_xor(ss, m, 32);
    if ((t & 31) == 0) red[t >> 5] = ss;
    __syncthreads();
    float tot = 0.0f;
#pragma unroll
    for (int i = 0; i < 8; ++i) tot += red[i];
    const float inv = rsqrtf(tot + EPS);
    v4bf o;
    o[0] = (__bf16)(x.x * inv);
    o[1] = (__bf16)(x.y * inv);
    o[2] = (__bf16)(x.z * inv);
    o[3] = (__bf16)(x.w * inv);
    ((v4bf*)(dst + (size_t)row * DIM))[t] = o;
}

// ---------------------------------------------------------------------------
// Per-row exact fp32 terms: sim_kk of back features -> pos/diagE, and pre_cos.
// One block per row.
// ---------------------------------------------------------------------------
__global__ __launch_bounds__(256)
void diag_pre_kernel(const float* __restrict__ pVF, const float* __restrict__ pAF,
                     const float* __restrict__ bVF, const float* __restrict__ bAF,
                     float* __restrict__ pos, float* __restrict__ diagE,
                     float* __restrict__ precos) {
    __shared__ float red[6][8];
    const int row = blockIdx.x;
    const int t   = threadIdx.x;
    const size_t base = (size_t)row * DIM;
    float4 a = ((const float4*)(pVF + base))[t];
    float4 b = ((const float4*)(pAF + base))[t];
    float4 c = ((const float4*)(bVF + base))[t];
    float4 d = ((const float4*)(bAF + base))[t];
    float acc[6];
    acc[0] = a.x * a.x + a.y * a.y + a.z * a.z + a.w * a.w;  // ||pV||^2
    acc[1] = b.x * b.x + b.y * b.y + b.z * b.z + b.w * b.w;  // ||pA||^2
    acc[2] = a.x * b.x + a.y * b.y + a.z * b.z + a.w * b.w;  // pV.pA
    acc[3] = c.x * c.x + c.y * c.y + c.z * c.z + c.w * c.w;  // ||bV||^2
    acc[4] = d.x * d.x + d.y * d.y + d.z * d.z + d.w * d.w;  // ||bA||^2
    acc[5] = c.x * d.x + c.y * d.y + c.z * d.z + c.w * d.w;  // bV.bA
#pragma unroll
    for (int j = 0; j < 6; ++j) {
        float v = acc[j];
#pragma unroll
        for (int m = 16; m >= 1; m >>= 1) v += __shfl_xor(v, m, 32);
        if ((t & 31) == 0) red[j][t >> 5] = v;
    }
    __syncthreads();
    if (t == 0) {
        float s[6];
#pragma unroll
        for (int j = 0; j < 6; ++j) {
            float v = 0.0f;
#pragma unroll
            for (int i = 0; i < 8; ++i) v += red[j][i];
            s[j] = v;
        }
        const float simKK  = s[5] * rsqrtf(s[3] + EPS) * rsqrtf(s[4] + EPS);
        const float preCos = s[2] * rsqrtf(s[0] + EPS) * rsqrtf(s[1] + EPS);
        pos[row]    = __expf(simKK - MARGIN);
        diagE[row]  = __expf(simKK);
        precos[row] = preCos;
    }
}

// ---------------------------------------------------------------------------
// WMMA GEMM: S = Vn @ An^T (bf16 operands, f32 accum), fused exp + row/col
// sum accumulation.
//   Block tile: 256(M) x 128(N), 8 waves arranged 4(M) x 2(N).
//   Wave tile:  64 x 64 = 4x4 of v_wmma_f32_16x16x32_bf16 per K-step
//               -> 16 WMMAs against 16 ds_load_b128 (1:1).
//   K-loop: 32 steps of K=32.
//   LDS rows padded to 40 bf16 (80 B) to break 64-bank conflicts.
// ---------------------------------------------------------------------------
__global__ __launch_bounds__(256, 1)
void sim_gemm_kernel(const __bf16* __restrict__ Vn, const __bf16* __restrict__ An,
                     float* __restrict__ rowsum, float* __restrict__ colsum) {
    __shared__ __bf16 As[256 * 40];   // 20 KB
    __shared__ __bf16 Bs[128 * 40];   // 10 KB

    const int t    = threadIdx.x;
    const int lane = t & 31;
    const int w    = t >> 5;
    const int lo   = lane & 15;
    const int hi   = lane >> 4;            // 0: lanes 0-15, 1: lanes 16-31
    const int m0   = blockIdx.y * 256;
    const int n0   = blockIdx.x * 128;
    const int wm   = (w & 3) * 64;         // wave M offset within block tile
    const int wn   = (w >> 2) * 64;        // wave N offset within block tile

    const v8f vzero = {0.f, 0.f, 0.f, 0.f, 0.f, 0.f, 0.f, 0.f};
    v8f acc[4][4];
#pragma unroll
    for (int ms = 0; ms < 4; ++ms)
#pragma unroll
        for (int ns = 0; ns < 4; ++ns) acc[ms][ns] = vzero;

    // Staging:
    //   A tile 256x32 bf16: thread t owns row t, loads all 64 B (4 x b128).
    //   B tile 128x32 bf16: thread t owns row t>>1, half-row 32 B (2 x b128).
    const int    br   = t >> 1;
    const int    bseg = (t & 1) * 16;      // bf16 element offset
    const size_t vgb  = (size_t)(m0 + t) * DIM;
    const size_t agb  = (size_t)(n0 + br) * DIM + bseg;

    for (int kk = 0; kk < DIM; kk += 32) {
        __syncthreads();
        const v8bf* gv = (const v8bf*)(Vn + vgb + kk);
        const v8bf* ga = (const v8bf*)(An + agb + kk);
        v8bf a0 = gv[0], a1 = gv[1], a2 = gv[2], a3 = gv[3];
        v8bf b0 = ga[0], b1 = ga[1];
        if (kk + 32 < DIM) {
            __builtin_prefetch((const void*)(Vn + vgb + kk + 32), 0, 1);
            __builtin_prefetch((const void*)(An + agb + kk + 32), 0, 1);
        }
        *(v8bf*)&As[t * 40]        = a0;
        *(v8bf*)&As[t * 40 + 8]    = a1;
        *(v8bf*)&As[t * 40 + 16]   = a2;
        *(v8bf*)&As[t * 40 + 24]   = a3;
        *(v8bf*)&Bs[br * 40 + bseg]     = b0;
        *(v8bf*)&Bs[br * 40 + bseg + 8] = b1;
        __syncthreads();

        // A fragment (16x32, 16-bit): lanes 0-15 -> K {0..7,16..23},
        // lanes 16-31 -> K {8..15,24..31}, one M-row per lane.
        union { v16bf f; v8bf h[2]; } afr[4], bfr[4];
#pragma unroll
        for (int ms = 0; ms < 4; ++ms) {
            const int off = (wm + ms * 16 + lo) * 40 + hi * 8;
            afr[ms].h[0] = *(const v8bf*)&As[off];
            afr[ms].h[1] = *(const v8bf*)&As[off + 16];
        }
        // B fragment (32x16, 16-bit): lanes 0-15 -> K {0..7,16..23},
        // lanes 16-31 -> K {8..15,24..31}, one N-column per lane.
#pragma unroll
        for (int ns = 0; ns < 4; ++ns) {
            const int off = (wn + ns * 16 + lo) * 40 + hi * 8;
            bfr[ns].h[0] = *(const v8bf*)&Bs[off];
            bfr[ns].h[1] = *(const v8bf*)&Bs[off + 16];
        }
#pragma unroll
        for (int ms = 0; ms < 4; ++ms)
#pragma unroll
            for (int ns = 0; ns < 4; ++ns)
                acc[ms][ns] = __builtin_amdgcn_wmma_f32_16x16x32_bf16(
                    false, afr[ms].f, false, bfr[ns].f,
                    (short)0, acc[ms][ns], false, false);
    }

    // Epilogue: E = exp(S); accumulate row sums (over N) and col sums (over M).
    // C/D layout: lane(0-15): M = vgpr, N = lane; lane(16-31): M = vgpr+8, N = lane-16.
    float rowpart[4][8];
#pragma unroll
    for (int ms = 0; ms < 4; ++ms)
#pragma unroll
        for (int v = 0; v < 8; ++v) rowpart[ms][v] = 0.0f;

#pragma unroll
    for (int ns = 0; ns < 4; ++ns) {
        float cs = 0.0f;
#pragma unroll
        for (int ms = 0; ms < 4; ++ms)
#pragma unroll
            for (int v = 0; v < 8; ++v) {
                const float e = __expf(acc[ms][ns][v]);
                rowpart[ms][v] += e;
                cs += e;
            }
        atomicAdd(&colsum[n0 + wn + ns * 16 + lo], cs);
    }

#pragma unroll
    for (int ms = 0; ms < 4; ++ms)
#pragma unroll
        for (int v = 0; v < 8; ++v) {
            float rv = rowpart[ms][v];
            rv += __shfl_xor(rv, 1, 32);
            rv += __shfl_xor(rv, 2, 32);
            rv += __shfl_xor(rv, 4, 32);
            rv += __shfl_xor(rv, 8, 32);
            if (lo == 0)
                atomicAdd(&rowsum[m0 + wm + ms * 16 + hi * 8 + v], rv);
        }
}

// ---------------------------------------------------------------------------
// Final scalar reduction (single block).
// ---------------------------------------------------------------------------
__global__ __launch_bounds__(256)
void final_kernel(const float* __restrict__ rowsum, const float* __restrict__ colsum,
                  const float* __restrict__ pos, const float* __restrict__ diagE,
                  const float* __restrict__ precos, float* __restrict__ out) {
    __shared__ float red[3][8];
    float sLV = 0.0f, sLA = 0.0f, sP = 0.0f;
    for (int i = threadIdx.x; i < NROWS; i += 256) {
        const float p  = pos[i];
        const float dE = diagE[i];
        sLV += __logf(p / (p + (rowsum[i] - dE)));
        sLA += __logf(p / (p + (colsum[i] - dE)));
        sP  += precos[i];
    }
    float vals[3] = {sLV, sLA, sP};
#pragma unroll
    for (int j = 0; j < 3; ++j) {
        float v = vals[j];
#pragma unroll
        for (int m = 16; m >= 1; m >>= 1) v += __shfl_xor(v, m, 32);
        if ((threadIdx.x & 31) == 0) red[j][threadIdx.x >> 5] = v;
    }
    __syncthreads();
    if (threadIdx.x == 0) {
        float s[3];
#pragma unroll
        for (int j = 0; j < 3; ++j) {
            float v = 0.0f;
#pragma unroll
            for (int i = 0; i < 8; ++i) v += red[j][i];
            s[j] = v;
        }
        // loss = 0.5 * (-1/1) * (L_V + L_A) + 0.5 * L_pre
        out[0] = BALANCE * (-1.0f) * (s[0] + s[1]) + (1.0f - BALANCE) * s[2];
    }
}

// ---------------------------------------------------------------------------
// Launcher.  Inputs (setup_inputs order): pre_VF, pre_AF, back_VF, back_AF.
// Workspace: Vn bf16 (8 MB) | An bf16 (8 MB) | rowsum | colsum | pos | diagE | precos
// ---------------------------------------------------------------------------
extern "C" void kernel_launch(void* const* d_in, const int* in_sizes, int n_in,
                              void* d_out, int out_size, void* d_ws, size_t ws_size,
                              hipStream_t stream) {
    const float* pre_VF  = (const float*)d_in[0];
    const float* pre_AF  = (const float*)d_in[1];
    const float* back_VF = (const float*)d_in[2];
    const float* back_AF = (const float*)d_in[3];
    float* out = (float*)d_out;

    char* ws = (char*)d_ws;
    __bf16* Vn = (__bf16*)(ws);
    __bf16* An = (__bf16*)(ws + (size_t)8 * 1024 * 1024);
    float* rowsum = (float*)(ws + (size_t)16 * 1024 * 1024);
    float* colsum = rowsum + NROWS;
    float* pos    = colsum + NROWS;
    float* diagE  = pos    + NROWS;
    float* precos = diagE  + NROWS;

    zero_kernel<<<(2 * NROWS + 255) / 256, 256, 0, stream>>>(rowsum, 2 * NROWS);
    norm_bf16_kernel<<<NROWS, 256, 0, stream>>>(back_VF, Vn);
    norm_bf16_kernel<<<NROWS, 256, 0, stream>>>(back_AF, An);
    diag_pre_kernel<<<NROWS, 256, 0, stream>>>(pre_VF, pre_AF, back_VF, back_AF,
                                               pos, diagE, precos);
    sim_gemm_kernel<<<dim3(NROWS / 128, NROWS / 256), 256, 0, stream>>>(Vn, An,
                                                                        rowsum, colsum);
    final_kernel<<<1, 256, 0, stream>>>(rowsum, colsum, pos, diagE, precos, out);
}
